// AttentionCommHU_52501680226760
// MI455X (gfx1250) — matrix operand order repaired and checked by reading, not compiled, not run on hardware
//
#include <hip/hip_runtime.h>
#include <hip/hip_bf16.h>

typedef __attribute__((ext_vector_type(16))) _Float16 v16h;
typedef __attribute__((ext_vector_type(8)))  _Float16 v8h;
typedef __attribute__((ext_vector_type(8)))  float    v8f;

#define N_AGENTS 8192
#define MEM_DIM  512
#define ENC_DIM  512
#define C_DIM    64
#define INV_SCALE 0.125f
#define EPS_RMS  1e-6f

__device__ __forceinline__ v8f wmma16(v16h a, v16h b, v8f c) {
  // D = A(16x32 f16) * B(32x16 f16) + C(16x16 f32)
  return __builtin_amdgcn_wmma_f32_16x16x32_f16(false, a, false, b, (short)0, c,
                                                false, false);
}

// A fragment (16 rows x 32 K) from f32 row-major src; also accumulates sum of
// squares of the touched elements (lane + lane^16 together cover a full row).
__device__ __forceinline__ v16h load_a_f32(const float* __restrict__ src, int ld,
                                           int r0, int k0, int ln, int hi,
                                           float* ssq) {
  const float* p = src + (size_t)(r0 + ln) * ld + k0 + 8 * hi;
  v16h a; float s = 0.f;
#pragma unroll
  for (int e = 0; e < 8; ++e) { float x = p[e];      s += x * x; a[e]     = (_Float16)x; }
#pragma unroll
  for (int e = 0; e < 8; ++e) { float x = p[16 + e]; s += x * x; a[8 + e] = (_Float16)x; }
  *ssq += s;
  return a;
}

// A fragment (16 rows x 32 K) from f16 row-major src.
__device__ __forceinline__ v16h load_a_h(const _Float16* __restrict__ M, int ld,
                                         int r0, int k0, int ln, int hi) {
  const _Float16* p = M + (size_t)(r0 + ln) * ld + k0 + 8 * hi;
  v16h a;
#pragma unroll
  for (int e = 0; e < 8; ++e) { a[e] = p[e]; a[8 + e] = p[16 + e]; }
  return a;
}

// B fragment (32 K x 16 N) where B[kk][n] = W[n0+n][k0+kk], W f32 row-major.
__device__ __forceinline__ v16h load_b_f32(const float* __restrict__ W, int ld,
                                           int n0, int k0, int ln, int hi) {
  const float* p = W + (size_t)(n0 + ln) * ld + k0 + 8 * hi;
  v16h b;
#pragma unroll
  for (int e = 0; e < 8; ++e) { b[e] = (_Float16)p[e]; b[8 + e] = (_Float16)p[16 + e]; }
  return b;
}

// B fragment (32 K x 16 N) where B[kk][n] = M[n0+n][k0+kk], M f16 row-major.
__device__ __forceinline__ v16h load_b_h(const _Float16* __restrict__ M, int ld,
                                         int n0, int k0, int ln, int hi) {
  const _Float16* p = M + (size_t)(n0 + ln) * ld + k0 + 8 * hi;
  v16h b;
#pragma unroll
  for (int e = 0; e < 8; ++e) { b[e] = p[e]; b[8 + e] = p[16 + e]; }
  return b;
}

// ---------------------------------------------------------------------------
// Kernel 1: q = RMS(RMS(h)@Wq^T + bq), k = RMS(u@Wk^T + bk), v = u@Wv^T + bv
// q,k stored f16 row-major [N,64]; v stored f16 TRANSPOSED [64,N] so the
// attention P@V WMMA gets contiguous B fragments.
// ---------------------------------------------------------------------------
__global__ __launch_bounds__(256) void proj_kernel(
    const float* __restrict__ h_query, const float* __restrict__ u_kv,
    const float* __restrict__ Wq, const float* __restrict__ bq,
    const float* __restrict__ Wk, const float* __restrict__ bk,
    const float* __restrict__ Wv, const float* __restrict__ bv,
    _Float16* __restrict__ q_h, _Float16* __restrict__ k_h,
    _Float16* __restrict__ v_t) {
  const int wid  = blockIdx.x * 8 + (threadIdx.x >> 5);
  const int lane = threadIdx.x & 31;
  const int ln = lane & 15, hi = lane >> 4;
  const int r0 = wid * 16;

  v8f accq[4], acck[4], accv[4];
#pragma unroll
  for (int t = 0; t < 4; ++t) { accq[t] = {}; acck[t] = {}; accv[t] = {}; }

  float ssh = 0.f, dum = 0.f;
  for (int c = 0; c < MEM_DIM; c += 32) {
    v16h ah = load_a_f32(h_query, MEM_DIM, r0, c, ln, hi, &ssh);
    v16h au = load_a_f32(u_kv,    ENC_DIM, r0, c, ln, hi, &dum);
#pragma unroll
    for (int t = 0; t < 4; ++t) {
      accq[t] = wmma16(ah, load_b_f32(Wq, MEM_DIM, 16 * t, c, ln, hi), accq[t]);
      acck[t] = wmma16(au, load_b_f32(Wk, ENC_DIM, 16 * t, c, ln, hi), acck[t]);
      accv[t] = wmma16(au, load_b_f32(Wv, ENC_DIM, 16 * t, c, ln, hi), accv[t]);
    }
  }

  // input RMS scale for h rows (rows live at lane index = row, both halves)
  ssh += __shfl_xor(ssh, 16, 32);
  float sc_h = rsqrtf(ssh * (1.0f / MEM_DIM) + EPS_RMS);
  float sh[8];
#pragma unroll
  for (int j = 0; j < 8; ++j) sh[j] = __shfl(sc_h, 8 * hi + j, 32);

  // ---- q path: scale rows, add bias, RMS over the 64 outputs ----
  float qpre[4][8]; float ssq[8];
#pragma unroll
  for (int j = 0; j < 8; ++j) ssq[j] = 0.f;
#pragma unroll
  for (int t = 0; t < 4; ++t) {
    float bias = bq[16 * t + ln];
#pragma unroll
    for (int j = 0; j < 8; ++j) {
      float v = accq[t][j] * sh[j] + bias;
      qpre[t][j] = v; ssq[j] += v * v;
    }
  }
#pragma unroll
  for (int j = 0; j < 8; ++j) {
#pragma unroll
    for (int off = 1; off < 16; off <<= 1) ssq[j] += __shfl_xor(ssq[j], off, 32);
    float rs = rsqrtf(ssq[j] * (1.0f / C_DIM) + EPS_RMS);
#pragma unroll
    for (int t = 0; t < 4; ++t)
      q_h[(size_t)(r0 + 8 * hi + j) * C_DIM + 16 * t + ln] =
          (_Float16)(qpre[t][j] * rs);
  }

  // ---- k path: bias + RMS ----
  float kpre[4][8]; float ssk[8];
#pragma unroll
  for (int j = 0; j < 8; ++j) ssk[j] = 0.f;
#pragma unroll
  for (int t = 0; t < 4; ++t) {
    float bias = bk[16 * t + ln];
#pragma unroll
    for (int j = 0; j < 8; ++j) {
      float v = acck[t][j] + bias;
      kpre[t][j] = v; ssk[j] += v * v;
    }
  }
#pragma unroll
  for (int j = 0; j < 8; ++j) {
#pragma unroll
    for (int off = 1; off < 16; off <<= 1) ssk[j] += __shfl_xor(ssk[j], off, 32);
    float rs = rsqrtf(ssk[j] * (1.0f / C_DIM) + EPS_RMS);
#pragma unroll
    for (int t = 0; t < 4; ++t)
      k_h[(size_t)(r0 + 8 * hi + j) * C_DIM + 16 * t + ln] =
          (_Float16)(kpre[t][j] * rs);
  }

  // ---- v path: bias, store transposed (16B packed stores) ----
#pragma unroll
  for (int t = 0; t < 4; ++t) {
    float bias = bv[16 * t + ln];
    v8h pack;
#pragma unroll
    for (int j = 0; j < 8; ++j) pack[j] = (_Float16)(accv[t][j] + bias);
    *(v8h*)(v_t + (size_t)(16 * t + ln) * N_AGENTS + r0 + 8 * hi) = pack;
  }
}

// ---------------------------------------------------------------------------
// Kernel 2: attention. One wave owns 16 query rows.
// Pass 1 (flash-style, no attn traffic): agg = sum exp(S)*v, rsum = sum exp(S).
// Pass 2: recompute S (cheap f16 WMMA), write attn = exp(S)/rsum once.
// |S| <= 8 because q,k are RMS-normalized -> no max subtraction needed.
// ---------------------------------------------------------------------------
__global__ __launch_bounds__(256) void attn_kernel(
    const _Float16* __restrict__ q_h, const _Float16* __restrict__ k_h,
    const _Float16* __restrict__ v_t, _Float16* __restrict__ agg_h,
    float* __restrict__ attn_out) {
  __shared__ _Float16 lds_p[8][16 * 32];   // per-wave 16x32 exp(S) tile
  const int wslot = threadIdx.x >> 5;
  const int wid   = blockIdx.x * 8 + wslot;
  const int lane  = threadIdx.x & 31;
  const int ln = lane & 15, hi = lane >> 4;
  const int r0 = wid * 16;

  v16h aq0 = load_a_h(q_h, C_DIM, r0, 0,  ln, hi);
  v16h aq1 = load_a_h(q_h, C_DIM, r0, 32, ln, hi);

  v8f agg[4];
#pragma unroll
  for (int t = 0; t < 4; ++t) agg[t] = {};
  float rsum[8];
#pragma unroll
  for (int j = 0; j < 8; ++j) rsum[j] = 0.f;

  for (int col0 = 0; col0 < N_AGENTS; col0 += 32) {
    if (col0 + 128 < N_AGENTS)
      __builtin_prefetch(k_h + (size_t)(col0 + 128) * C_DIM, 0, 1);
#pragma unroll
    for (int t2 = 0; t2 < 2; ++t2) {
      const int n0 = col0 + 16 * t2;
      v8f s = {};
      s = wmma16(aq0, load_b_h(k_h, C_DIM, n0, 0,  ln, hi), s);
      s = wmma16(aq1, load_b_h(k_h, C_DIM, n0, 32, ln, hi), s);
      const int col = n0 + ln;
#pragma unroll
      for (int j = 0; j < 8; ++j) {
        const int row = r0 + 8 * hi + j;
        float e = (row == col) ? 0.f : __expf(s[j] * INV_SCALE);
        rsum[j] += e;
        lds_p[wslot][(8 * hi + j) * 32 + 16 * t2 + ln] = (_Float16)e;  // ds_store
      }
    }
    __syncthreads();
    v16h ap;  // re-read exp tile in A layout (rows striped within lane)
#pragma unroll
    for (int e = 0; e < 8; ++e) {
      ap[e]     = lds_p[wslot][ln * 32 + 8 * hi + e];
      ap[8 + e] = lds_p[wslot][ln * 32 + 8 * hi + 16 + e];
    }
#pragma unroll
    for (int t = 0; t < 4; ++t)
      agg[t] = wmma16(ap, load_b_h(v_t, N_AGENTS, 16 * t, col0, ln, hi), agg[t]);
    __syncthreads();
  }

  float inv[8];
#pragma unroll
  for (int j = 0; j < 8; ++j) {
#pragma unroll
    for (int off = 1; off < 16; off <<= 1) rsum[j] += __shfl_xor(rsum[j], off, 32);
    inv[j] = 1.0f / rsum[j];
  }
#pragma unroll
  for (int t = 0; t < 4; ++t)
#pragma unroll
    for (int j = 0; j < 8; ++j)
      agg_h[(size_t)(r0 + 8 * hi + j) * C_DIM + 16 * t + ln] =
          (_Float16)(agg[t][j] * inv[j]);

  // ---- pass 2: recompute scores, stream normalized attn (write-only) ----
  for (int col0 = 0; col0 < N_AGENTS; col0 += 16) {
    v8f s = {};
    s = wmma16(aq0, load_b_h(k_h, C_DIM, col0, 0,  ln, hi), s);
    s = wmma16(aq1, load_b_h(k_h, C_DIM, col0, 32, ln, hi), s);
    const int col = col0 + ln;
#pragma unroll
    for (int j = 0; j < 8; ++j) {
      const int row = r0 + 8 * hi + j;
      float e = (row == col) ? 0.f : __expf(s[j] * INV_SCALE) * inv[j];
      attn_out[(size_t)row * N_AGENTS + col] = e;
    }
  }
}

// ---------------------------------------------------------------------------
// Kernel 3: c = agg @ Wo^T + bo   ([8192,64] x [64,512])
// ---------------------------------------------------------------------------
__global__ __launch_bounds__(256) void out_kernel(
    const _Float16* __restrict__ agg_h, const float* __restrict__ Wo,
    const float* __restrict__ bo, float* __restrict__ c_out) {
  const int wid  = blockIdx.x * 8 + (threadIdx.x >> 5);
  const int lane = threadIdx.x & 31;
  const int ln = lane & 15, hi = lane >> 4;
  const int r0 = wid * 16;

  v16h a0 = load_a_h(agg_h, C_DIM, r0, 0,  ln, hi);
  v16h a1 = load_a_h(agg_h, C_DIM, r0, 32, ln, hi);

  for (int n0 = 0; n0 < ENC_DIM; n0 += 16) {
    v8f acc = {};
    acc = wmma16(a0, load_b_f32(Wo, C_DIM, n0, 0,  ln, hi), acc);
    acc = wmma16(a1, load_b_f32(Wo, C_DIM, n0, 32, ln, hi), acc);
    const float bias = bo[n0 + ln];
#pragma unroll
    for (int j = 0; j < 8; ++j)
      c_out[(size_t)(r0 + 8 * hi + j) * ENC_DIM + n0 + ln] = acc[j] + bias;
  }
}

extern "C" void kernel_launch(void* const* d_in, const int* in_sizes, int n_in,
                              void* d_out, int out_size, void* d_ws, size_t ws_size,
                              hipStream_t stream) {
  const float* h_query = (const float*)d_in[0];
  const float* u_kv    = (const float*)d_in[1];
  const float* Wq = (const float*)d_in[2];
  const float* bq = (const float*)d_in[3];
  const float* Wk = (const float*)d_in[4];
  const float* bk = (const float*)d_in[5];
  const float* Wv = (const float*)d_in[6];
  const float* bv = (const float*)d_in[7];
  const float* Wo = (const float*)d_in[8];
  const float* bo = (const float*)d_in[9];

  float* c_out    = (float*)d_out;
  float* attn_out = c_out + (size_t)N_AGENTS * ENC_DIM;

  _Float16* q_h   = (_Float16*)d_ws;                     // [N,64]
  _Float16* k_h   = q_h + (size_t)N_AGENTS * C_DIM;      // [N,64]
  _Float16* v_t   = k_h + (size_t)N_AGENTS * C_DIM;      // [64,N] transposed
  _Float16* agg_h = v_t + (size_t)C_DIM * N_AGENTS;      // [N,64]

  dim3 blk(256);
  dim3 grd(N_AGENTS / 128);  // 8 waves/block, 16 rows/wave -> 512 waves

  proj_kernel<<<grd, blk, 0, stream>>>(h_query, u_kv, Wq, bq, Wk, bk, Wv, bv,
                                       q_h, k_h, v_t);
  attn_kernel<<<grd, blk, 0, stream>>>(q_h, k_h, v_t, agg_h, attn_out);
  out_kernel<<<grd, blk, 0, stream>>>(agg_h, Wo, bo, c_out);
}
